// CopyTokenDecoder_33517924778139
// MI455X (gfx1250) — matrix operand order repaired and checked by reading, not compiled, not run on hardware
//
#include <hip/hip_runtime.h>
#include <hip/hip_bf16.h>
#include <math.h>

typedef __bf16 bf16;
typedef __attribute__((ext_vector_type(16))) __bf16 v16bf;
typedef __attribute__((ext_vector_type(8)))  __bf16 v8bf;
typedef __attribute__((ext_vector_type(8)))  float  v8f;

#define TPB 256

// ---------------------------------------------------------------- elementwise
__global__ __launch_bounds__(TPB) void cast_f32_bf16_kernel(
    const float* __restrict__ src, bf16* __restrict__ dst, long long n) {
  long long i = (long long)blockIdx.x * TPB + threadIdx.x;
  if (i < n) dst[i] = (bf16)src[i];
}

// v (S,B,D) f32 -> vt (B,D,S) bf16
__global__ __launch_bounds__(TPB) void transpose_v_kernel(
    const float* __restrict__ v, bf16* __restrict__ vt, int S, int B, int D) {
  long long idx = (long long)blockIdx.x * TPB + threadIdx.x;
  long long n = (long long)S * B * D;
  if (idx >= n) return;
  int s = (int)(idx % S);
  long long r = idx / S;
  int d = (int)(r % D);
  int b = (int)(r / D);
  vt[idx] = (bf16)v[((long long)s * B + b) * D + d];
}

// ---------------------------------------------------------------- WMMA GEMM
// D(M,N) = scale * (A(M,K) @ W(N,K)^T + bias[n]), optional relu.
// Writes f32 to C (if non-null) and/or bf16 to Cbf (if non-null).
// A, W bf16 row-major (row strides lda/ldb); output row stride ldc.
// blockIdx.z batches via element strides abat/bbat/cbat.
// Block = 256 threads = 8 waves (2 x 4); block tile 128(M) x 256(N);
// wave tile 64x64 = 4x4 grid of 16x16 WMMA tiles (16 wmma per 32-K step).
__global__ __launch_bounds__(TPB) void gemm_bf16_kernel(
    float* __restrict__ C, bf16* __restrict__ Cbf,
    const bf16* __restrict__ A, const bf16* __restrict__ W,
    const float* __restrict__ bias, int M, int N, int K,
    int lda, int ldb, int ldc,
    long long abat, long long bbat, long long cbat,
    float scale, int relu) {
  int z = blockIdx.z;
  A += (long long)z * abat;
  W += (long long)z * bbat;
  if (C)   C   += (long long)z * cbat;
  if (Cbf) Cbf += (long long)z * cbat;

  int tid = threadIdx.x;
  int lane = tid & 31, wave = tid >> 5;
  int wm = wave >> 2;        // 0..1
  int wn = wave & 3;         // 0..3
  int m0 = blockIdx.y * 128 + wm * 64;
  int n0 = blockIdx.x * 256 + wn * 64;
  if (m0 >= M || n0 >= N) return;

  int hl = lane >> 4;        // 0 for lanes 0-15, 1 for lanes 16-31
  int ln = lane & 15;

  v8f zero = {0.f, 0.f, 0.f, 0.f, 0.f, 0.f, 0.f, 0.f};
  v8f acc[4][4];
#pragma unroll
  for (int mi = 0; mi < 4; ++mi)
#pragma unroll
    for (int ni = 0; ni < 4; ++ni) acc[mi][ni] = zero;

  const bf16* arow[4];
  const bf16* brow[4];
#pragma unroll
  for (int mi = 0; mi < 4; ++mi)
    arow[mi] = A + (long long)(m0 + mi * 16 + ln) * lda;
#pragma unroll
  for (int ni = 0; ni < 4; ++ni)
    brow[ni] = W + (long long)(n0 + ni * 16 + ln) * ldb;

  for (int kk = 0; kk < K; kk += 32) {
    // A fragments 16x32: elems 0..15 = K[hl*16..+16)
    v16bf af[4];
#pragma unroll
    for (int mi = 0; mi < 4; ++mi) {
      v8bf lo = *(const v8bf*)(arow[mi] + kk + hl * 16);
      v8bf hi = *(const v8bf*)(arow[mi] + kk + hl * 16 + 8);
#pragma unroll
      for (int i = 0; i < 8; ++i) { af[mi][i] = lo[i]; af[mi][i + 8] = hi[i]; }
    }
    // B fragments 32x16: lane holds col n, elems j = W[n][kk + hl*16 + j]
    v16bf bfr[4];
#pragma unroll
    for (int ni = 0; ni < 4; ++ni)
      bfr[ni] = *(const v16bf*)(brow[ni] + kk + hl * 16);
#pragma unroll
    for (int mi = 0; mi < 4; ++mi)
#pragma unroll
      for (int ni = 0; ni < 4; ++ni)
        acc[mi][ni] = __builtin_amdgcn_wmma_f32_16x16x32_bf16(
            false, af[mi], false, bfr[ni], (short)0, acc[mi][ni], false, false);
  }

#pragma unroll
  for (int mi = 0; mi < 4; ++mi) {
#pragma unroll
    for (int ni = 0; ni < 4; ++ni) {
      int col = n0 + ni * 16 + ln;
      float bv = bias ? bias[col] : 0.f;
#pragma unroll
      for (int r = 0; r < 8; ++r) {
        int row = m0 + mi * 16 + hl * 8 + r;  // C layout: VGPR r -> row hl*8+r
        float val = (acc[mi][ni][r] + bv) * scale;
        if (relu) val = fmaxf(val, 0.f);
        long long idx = (long long)row * ldc + col;
        if (C)   C[idx] = val;
        if (Cbf) Cbf[idx] = (bf16)val;
      }
    }
  }
}

// ---------------------------------------------------------------- reductions
__device__ inline float block_sum256(float v, float* sm) {
  int tid = threadIdx.x;
  sm[tid] = v; __syncthreads();
  for (int s = 128; s > 0; s >>= 1) {
    if (tid < s) sm[tid] += sm[tid + s];
    __syncthreads();
  }
  float r = sm[0]; __syncthreads();
  return r;
}
__device__ inline float block_max256(float v, float* sm) {
  int tid = threadIdx.x;
  sm[tid] = v; __syncthreads();
  for (int s = 128; s > 0; s >>= 1) {
    if (tid < s) sm[tid] = fmaxf(sm[tid], sm[tid + s]);
    __syncthreads();
  }
  float r = sm[0]; __syncthreads();
  return r;
}

// softmax over S with mask; writes f32 aw in place + bf16 copy. One block per (b,t).
__global__ __launch_bounds__(TPB) void softmax_scores_kernel(
    float* __restrict__ scores, bf16* __restrict__ aw_bf,
    const unsigned char* __restrict__ mask, int T, int S) {
  __shared__ float sm[TPB];
  int row = blockIdx.x;            // b*T + t
  int b = row / T;
  float* x = scores + (long long)row * S;
  int tid = threadIdx.x;
  float mx = -3.4e38f;
  for (int s = tid; s < S; s += TPB) {
    float v = mask[b * S + s] ? -1e30f : x[s];
    x[s] = v;
    mx = fmaxf(mx, v);
  }
  mx = block_max256(mx, sm);
  float sum = 0.f;
  for (int s = tid; s < S; s += TPB) {
    float e = __expf(x[s] - mx);
    x[s] = e;
    sum += e;
  }
  sum = block_sum256(sum, sm);
  float inv = 1.f / sum;
  for (int s = tid; s < S; s += TPB) {
    float p = x[s] * inv;
    x[s] = p;
    aw_bf[(long long)row * S + s] = (bf16)p;
  }
}

// Per row (t*B+b), D=512, 256 threads (2 elems/thread):
//   attn_n = LN(attn, ln1); gates = softmax([outs, attn_n] @ Wd.T + bd)
//   x = LN(outs + attn, ln1) -> bf16
__global__ __launch_bounds__(TPB) void gate_ln_kernel(
    const float* __restrict__ outs, const float* __restrict__ attn,
    const float* __restrict__ g1, const float* __restrict__ b1,
    const float* __restrict__ Wd, const float* __restrict__ bd,
    float* __restrict__ gates, bf16* __restrict__ x_bf, int D) {
  __shared__ float sm[TPB];
  int row = blockIdx.x, tid = threadIdx.x;
  const float* o = outs + (long long)row * D;
  const float* a = attn + (long long)row * D;
  int i0 = tid, i1 = tid + TPB;
  float a0 = a[i0], a1 = a[i1];
  float o0 = o[i0], o1 = o[i1];

  // LN(attn)
  float mu = block_sum256(a0 + a1, sm) / D;
  float var = block_sum256((a0 - mu) * (a0 - mu) + (a1 - mu) * (a1 - mu), sm) / D;
  float rstd = rsqrtf(var + 1e-5f);
  float an0 = (a0 - mu) * rstd * g1[i0] + b1[i0];
  float an1 = (a1 - mu) * rstd * g1[i1] + b1[i1];

  // gate logits: Wd row0 = [outs | attn_n] weights, row1 at offset 2D
  float d0 = o0 * Wd[i0] + o1 * Wd[i1] + an0 * Wd[D + i0] + an1 * Wd[D + i1];
  float d1 = o0 * Wd[2 * D + i0] + o1 * Wd[2 * D + i1] +
             an0 * Wd[3 * D + i0] + an1 * Wd[3 * D + i1];
  d0 = block_sum256(d0, sm);
  d1 = block_sum256(d1, sm);
  if (tid == 0) {
    d0 += bd[0]; d1 += bd[1];
    float m = fmaxf(d0, d1);
    float e0 = __expf(d0 - m), e1 = __expf(d1 - m);
    float inv = 1.f / (e0 + e1);
    gates[row * 2 + 0] = e0 * inv;
    gates[row * 2 + 1] = e1 * inv;
  }

  // x = LN(outs + attn)
  float r0 = o0 + a0, r1 = o1 + a1;
  float rmu = block_sum256(r0 + r1, sm) / D;
  float rvar = block_sum256((r0 - rmu) * (r0 - rmu) + (r1 - rmu) * (r1 - rmu), sm) / D;
  float rrstd = rsqrtf(rvar + 1e-5f);
  x_bf[(long long)row * D + i0] = (bf16)((r0 - rmu) * rrstd * g1[i0] + b1[i0]);
  x_bf[(long long)row * D + i1] = (bf16)((r1 - rmu) * rrstd * g1[i1] + b1[i1]);
}

// x2 = LN(h, ln2) -> bf16; one block per row, D=512
__global__ __launch_bounds__(TPB) void ln2_kernel(
    const float* __restrict__ x, const float* __restrict__ g,
    const float* __restrict__ b, bf16* __restrict__ out, int D) {
  __shared__ float sm[TPB];
  int row = blockIdx.x, tid = threadIdx.x;
  const float* xr = x + (long long)row * D;
  int i0 = tid, i1 = tid + TPB;
  float v0 = xr[i0], v1 = xr[i1];
  float mu = block_sum256(v0 + v1, sm) / D;
  float var = block_sum256((v0 - mu) * (v0 - mu) + (v1 - mu) * (v1 - mu), sm) / D;
  float rstd = rsqrtf(var + 1e-5f);
  out[(long long)row * D + i0] = (bf16)((v0 - mu) * rstd * g[i0] + b[i0]);
  out[(long long)row * D + i1] = (bf16)((v1 - mu) * rstd * g[i1] + b[i1]);
}

// Fused epilogue per (t,b) row of d_out (V logits), minimal HBM passes:
//  pass 1: online (max,sum) reduction over logits           (1 read)
//  pass 2: write log(gen*softmax + 1e-12) for every j       (1 read + 1 write)
//  fixup : serially apply <=512 copy-scatter adds by undoing the log on the
//          affected ids only (deterministic order, matches graph replays)
__global__ __launch_bounds__(TPB) void vocab_kernel(
    float* __restrict__ out, const float* __restrict__ gates,
    const float* __restrict__ aw, const int* __restrict__ copy_seq,
    int T, int B, int S, int V) {
  __shared__ float smx[TPB], ssm[TPB];
  __shared__ int   svid[512];
  __shared__ float sval[512];
  int row = blockIdx.x;            // t*B + b
  int t = row / B, b = row % B;
  float* p = out + (long long)row * V;
  int tid = threadIdx.x;

  // pass 1: per-thread online softmax stats, then pairwise block combine
  float m = -3.4e38f, s = 0.f;
  for (int j = tid; j < V; j += TPB) {
    float v = p[j];
    float nm = fmaxf(m, v);
    s = s * __expf(m - nm) + __expf(v - nm);
    m = nm;
  }
  smx[tid] = m; ssm[tid] = s;
  __syncthreads();
  for (int off = 128; off > 0; off >>= 1) {
    if (tid < off) {
      float m2 = smx[tid + off], s2 = ssm[tid + off];
      float nm = fmaxf(smx[tid], m2);
      ssm[tid] = ssm[tid] * __expf(smx[tid] - nm) + s2 * __expf(m2 - nm);
      smx[tid] = nm;
    }
    __syncthreads();
  }
  float M_ = smx[0], S_ = ssm[0];
  __syncthreads();

  float gen = gates[row * 2 + 0], cg = gates[row * 2 + 1];
  float sc = gen / S_;

  // stage copy contributions while pass 2 streams
  for (int si = tid; si < S; si += TPB) {
    svid[si] = copy_seq[si * B + b];
    sval[si] = cg * aw[((long long)b * T + t) * S + si];
  }

  // pass 2: fused normalize + log
  for (int j = tid; j < V; j += TPB)
    p[j] = logf(__expf(p[j] - M_) * sc + 1e-12f);
  __syncthreads();

  // fixup affected vocab ids (serial -> deterministic float order)
  if (tid == 0) {
    for (int si = 0; si < S; ++si) {
      int id = svid[si];
      float pr = __expf(p[id]);   // = prob + 1e-12
      pr += sval[si];
      p[id] = logf(pr);
    }
  }
}

// ---------------------------------------------------------------- host
extern "C" void kernel_launch(void* const* d_in, const int* in_sizes, int n_in,
                              void* d_out, int out_size, void* d_ws, size_t ws_size,
                              hipStream_t stream) {
  (void)in_sizes; (void)n_in; (void)out_size; (void)ws_size;
  const int T = 256, B = 8, S = 512, D = 512, F = 2048, V = 32000;
  const long long TB = (long long)T * B;   // 2048 rows
  const long long SB = (long long)S * B;   // 4096 rows

  const float* outs  = (const float*)d_in[0];
  const float* mem   = (const float*)d_in[1];
  const unsigned char* mem_mask = (const unsigned char*)d_in[2];
  const int*   copy_seq = (const int*)d_in[3];
  const float* Wq = (const float*)d_in[4];  const float* bq = (const float*)d_in[5];
  const float* Wk = (const float*)d_in[6];  const float* bk = (const float*)d_in[7];
  const float* Wv = (const float*)d_in[8];  const float* bv = (const float*)d_in[9];
  const float* Wo = (const float*)d_in[10]; const float* bo = (const float*)d_in[11];
  const float* ln1_g = (const float*)d_in[12]; const float* ln1_b = (const float*)d_in[13];
  const float* Wd = (const float*)d_in[14]; const float* bd = (const float*)d_in[15];
  const float* W1 = (const float*)d_in[16]; const float* b1 = (const float*)d_in[17];
  const float* W2 = (const float*)d_in[18]; const float* b2 = (const float*)d_in[19];
  const float* ln2_g = (const float*)d_in[20]; const float* ln2_b = (const float*)d_in[21];
  const float* W_emb = (const float*)d_in[22];

  char* ws = (char*)d_ws;
  size_t off = 0;
  auto alloc = [&](long long bytes) -> void* {
    void* p = ws + off;
    off = (off + (size_t)bytes + 255) & ~(size_t)255;
    return p;
  };

  bf16* outs_bf = (bf16*)alloc(TB * D * 2);
  bf16* mem_bf  = (bf16*)alloc(SB * D * 2);
  bf16* Wq_bf = (bf16*)alloc((long long)D * D * 2);
  bf16* Wk_bf = (bf16*)alloc((long long)D * D * 2);
  bf16* Wv_bf = (bf16*)alloc((long long)D * D * 2);
  bf16* Wo_bf = (bf16*)alloc((long long)D * D * 2);
  bf16* W1_bf = (bf16*)alloc((long long)F * D * 2);
  bf16* W2_bf = (bf16*)alloc((long long)D * F * 2);
  bf16* We_bf = (bf16*)alloc((long long)V * D * 2);
  bf16*  q_bf = (bf16*)alloc(TB * D * 2);
  bf16*  k_bf = (bf16*)alloc(SB * D * 2);
  float* vf = (float*)alloc(SB * D * 4);
  bf16*  vt_bf = (bf16*)alloc((long long)B * D * S * 2);
  float* aw = (float*)alloc((long long)B * T * S * 4);
  bf16*  aw_bf = (bf16*)alloc((long long)B * T * S * 2);
  bf16*  attn0_bf = (bf16*)alloc(TB * D * 2);
  float* attnf = (float*)alloc(TB * D * 4);
  float* gates = (float*)alloc(TB * 2 * 4);
  bf16*  x_bf = (bf16*)alloc(TB * D * 2);
  bf16*  h_bf = (bf16*)alloc(TB * F * 2);
  float* x2f = (float*)alloc(TB * D * 4);
  bf16*  x2_bf = (bf16*)alloc(TB * D * 2);

  auto cast = [&](const float* s, bf16* d, long long n) {
    cast_f32_bf16_kernel<<<dim3((unsigned)((n + TPB - 1) / TPB)), dim3(TPB), 0, stream>>>(s, d, n);
  };
  auto gemm = [&](float* C, bf16* Cbf, const bf16* A, const bf16* Bm, const float* bias,
                  int M, int N, int K, int lda, int ldb, int ldc,
                  long long abat, long long bbat, long long cbat, int nbat,
                  float scale, int relu) {
    dim3 g((unsigned)(N / 256), (unsigned)(M / 128), (unsigned)nbat);
    gemm_bf16_kernel<<<g, dim3(TPB), 0, stream>>>(C, Cbf, A, Bm, bias, M, N, K,
                                                  lda, ldb, ldc, abat, bbat, cbat,
                                                  scale, relu);
  };

  // ---- bf16 conversions
  cast(outs, outs_bf, TB * D);
  cast(mem, mem_bf, SB * D);
  cast(Wq, Wq_bf, (long long)D * D);
  cast(Wk, Wk_bf, (long long)D * D);
  cast(Wv, Wv_bf, (long long)D * D);
  cast(Wo, Wo_bf, (long long)D * D);
  cast(W1, W1_bf, (long long)F * D);
  cast(W2, W2_bf, (long long)D * F);
  cast(W_emb, We_bf, (long long)V * D);

  // ---- q/k/v projections (q, k straight to bf16)
  float qscale = 1.0f / sqrtf((float)D);
  gemm(nullptr, q_bf, outs_bf, Wq_bf, bq, (int)TB, D, D, D, D, D, 0, 0, 0, 1, qscale, 0);
  gemm(nullptr, k_bf, mem_bf, Wk_bf, bk, (int)SB, D, D, D, D, D, 0, 0, 0, 1, 1.0f, 0);
  gemm(vf, nullptr, mem_bf, Wv_bf, bv, (int)SB, D, D, D, D, D, 0, 0, 0, 1, 1.0f, 0);
  {
    long long n = (long long)B * D * S;
    transpose_v_kernel<<<dim3((unsigned)((n + TPB - 1) / TPB)), dim3(TPB), 0, stream>>>(
        vf, vt_bf, S, B, D);
  }

  // ---- scores[b] = q_b (T,D) @ k_b (S,D)^T  (batched via strides)
  gemm(aw, nullptr, q_bf, k_bf, nullptr, T, S, D,
       B * D, B * D, S, (long long)D, (long long)D, (long long)T * S, B, 1.0f, 0);
  softmax_scores_kernel<<<dim3((unsigned)(B * T)), dim3(TPB), 0, stream>>>(
      aw, aw_bf, mem_mask, T, S);

  // ---- attn0[b] = aw_b (T,S) @ vt_b (D,S)^T, written (T,B,D)-interleaved bf16
  gemm(nullptr, attn0_bf, aw_bf, vt_bf, nullptr, T, D, S,
       S, S, B * D, (long long)T * S, (long long)D * S, (long long)D, B, 1.0f, 0);
  gemm(attnf, nullptr, attn0_bf, Wo_bf, bo, (int)TB, D, D, D, D, D, 0, 0, 0, 1, 1.0f, 0);

  // ---- gates + residual LN -> x_bf
  gate_ln_kernel<<<dim3((unsigned)TB), dim3(TPB), 0, stream>>>(
      outs, attnf, ln1_g, ln1_b, Wd, bd, gates, x_bf, D);

  // ---- FFN (h straight to bf16 with relu)
  gemm(nullptr, h_bf, x_bf, W1_bf, b1, (int)TB, F, D, D, D, F, 0, 0, 0, 1, 1.0f, 1);
  gemm(x2f, nullptr, h_bf, W2_bf, b2, (int)TB, D, F, F, F, D, 0, 0, 0, 1, 1.0f, 0);
  ln2_kernel<<<dim3((unsigned)TB), dim3(TPB), 0, stream>>>(x2f, ln2_g, ln2_b, x2_bf, D);

  // ---- vocab logits straight into d_out, then fused online-softmax/copy/log
  gemm((float*)d_out, nullptr, x2_bf, We_bf, nullptr, (int)TB, V, D, D, D, V, 0, 0, 0, 1, 1.0f, 0);
  vocab_kernel<<<dim3((unsigned)TB), dim3(TPB), 0, stream>>>(
      (float*)d_out, gates, aw, copy_seq, T, B, S, V);
}
